// EGNNLayer_4252017623571
// MI455X (gfx1250) — matrix operand order repaired and checked by reading, not compiled, not run on hardware
//
#include <hip/hip_runtime.h>
#include <hip/hip_bf16.h>

#define BB 8
#define NNODES 256
#define HH 128

typedef __attribute__((ext_vector_type(16))) __bf16 v16bf;
typedef __attribute__((ext_vector_type(8)))  float  v8f;

// ---------------------------------------------------------------------------
// WMMA helpers (CDNA5 v_wmma_f32_16x16x32_bf16)
// ---------------------------------------------------------------------------
__device__ __forceinline__ v8f wmma_bf16(v16bf a, v16bf b, v8f c) {
    return __builtin_amdgcn_wmma_f32_16x16x32_bf16(
        /*neg_a=*/false, a, /*neg_b=*/false, b,
        /*c_mod=*/(short)0, c, /*reuse_a=*/false, /*reuse_b=*/false);
}

// A-matrix 16x32 bf16 fragment from a row-major bf16 tile in LDS.
// Lane l (m = l&15, half = l>>4) holds row m with
//   elements 0..7  -> k = kt*32 + 8*half + [0..7]
//   elements 8..15 -> k = kt*32 + 16 + 8*half + [0..7]
// => two contiguous 16-byte chunks (2x ds_load_b128).
__device__ __forceinline__ v16bf load_a_frag(const __bf16* tile, int stride, int kt, int lane) {
    int m  = lane & 15;
    int hf = lane >> 4;
    const __bf16* p = tile + m * stride + kt * 32 + hf * 8;
    union { uint4 q[2]; v16bf v; } u;
    u.q[0] = *(const uint4*)(p);
    u.q[1] = *(const uint4*)(p + 16);
    return u.v;
}

// B fragments are pre-packed lane-linear: frag*512 + lane*16 bf16.
__device__ __forceinline__ v16bf load_b_frag(const __bf16* pk, int frag, int lane) {
    return *(const v16bf*)(pk + (frag << 9) + (lane << 4));
}

__device__ __forceinline__ float silu_f(float x) { return x / (1.0f + __expf(-x)); }

// ---------------------------------------------------------------------------
// Kernel 1: pack a row-major fp32 weight [K][N] into bf16 B-fragment layout.
// B-matrix 32x16 bf16: lane l holds column n = nt*16 + (l&15),
// k_local = 8*(l>>4) + e for e<8, 16 + 8*(l>>4) + (e-8) for e>=8.
// Fragment order: [nt][kt] (kt innermost for streaming in the K loop).
// ---------------------------------------------------------------------------
__global__ void pack_w_kernel(const float* __restrict__ W, __bf16* __restrict__ dst,
                              int K, int N) {
    int KT = K >> 5, NT = N >> 4;
    int total = KT * NT * 512;
    for (int idx = blockIdx.x * blockDim.x + threadIdx.x; idx < total;
         idx += gridDim.x * blockDim.x) {
        int e    = idx & 15;
        int lane = (idx >> 4) & 31;
        int frag = idx >> 9;
        int kt   = frag % KT;
        int nt   = frag / KT;
        int n = (nt << 4) + (lane & 15);
        int k = (kt << 5) + ((lane >> 4) << 3) + (e & 7) + ((e >> 3) << 4);
        dst[idx] = (__bf16)W[k * N + n];
    }
}

// ---------------------------------------------------------------------------
// Kernel 2: P = h @ W1a, Q = h @ W1b  (WMMA, bf16 in / f32 out)
// grid = B * N/16 workgroups, 256 threads (8 waves), wave w owns nt=w.
// ---------------------------------------------------------------------------
__global__ void pq_kernel(const float* __restrict__ h,
                          const __bf16* __restrict__ pkA,
                          const __bf16* __restrict__ pkB,
                          float* __restrict__ P, float* __restrict__ Q) {
    extern __shared__ char smem_raw[];
    __bf16* s_h = (__bf16*)smem_raw;        // [16][128] bf16 = 4 KB
    int b  = blockIdx.x >> 4;
    int i0 = (blockIdx.x & 15) << 4;
    int tid = threadIdx.x, lane = tid & 31, wave = tid >> 5;

    for (int x = tid; x < 16 * HH; x += blockDim.x) {
        int m = x >> 7, c = x & 127;
        s_h[x] = (__bf16)h[((b * NNODES) + i0 + m) * HH + c];
    }
    __syncthreads();

    int nt = wave;
    v8f accP, accQ;
    for (int r = 0; r < 8; r++) { accP[r] = 0.f; accQ[r] = 0.f; }
    for (int kt = 0; kt < 4; kt++) {
        v16bf a = load_a_frag(s_h, HH, kt, lane);
        accP = wmma_bf16(a, load_b_frag(pkA, nt * 4 + kt, lane), accP);
        accQ = wmma_bf16(a, load_b_frag(pkB, nt * 4 + kt, lane), accQ);
    }
    int col = lane & 15, hf = lane >> 4;
    for (int r = 0; r < 8; r++) {
        int m = r + 8 * hf;
        int n = nt * 16 + col;
        P[((b * NNODES) + i0 + m) * HH + n] = accP[r];
        Q[((b * NNODES) + i0 + m) * HH + n] = accQ[r];
    }
}

// ---------------------------------------------------------------------------
// Kernel 3: fused edge MLP + attention + coord update + node MLP + LayerNorm.
// grid = B * N/16 workgroups; 256 threads (8 waves). Workgroup owns a 16-row
// i-tile; wave w handles j in [w*32, w*32+32).
// ---------------------------------------------------------------------------
__global__ void egnn_edge_kernel(
    const float* __restrict__ h, const float* __restrict__ frac,
    const float* __restrict__ latt, const float* __restrict__ maskg,
    const float* __restrict__ b_m1, const float* __restrict__ b_m2,
    const float* __restrict__ b_c1, const float* __restrict__ Wc2,
    const float* __restrict__ b_c2, const float* __restrict__ b_n1,
    const float* __restrict__ b_n2, const float* __restrict__ ln_g,
    const float* __restrict__ ln_b,
    const __bf16* __restrict__ pkW1c, const __bf16* __restrict__ pkWm2,
    const __bf16* __restrict__ pkWc1, const __bf16* __restrict__ pkWn1,
    const __bf16* __restrict__ pkWn2,
    const float* __restrict__ P, const float* __restrict__ Q,
    float* __restrict__ out_h, float* __restrict__ out_coord)
{
    extern __shared__ char smem_raw[];
    __bf16* s_w1c   = (__bf16*)(smem_raw + 0);       //  8192 B (8 frags)
    __bf16* s_wm2   = (__bf16*)(smem_raw + 8192);    // 32768 B (32 frags)
    __bf16* s_wc1   = (__bf16*)(smem_raw + 40960);   // 32768 B
    float*  s_P     = (float*)(smem_raw + 73728);    //  8192 B [16][128] (+b1)
    __bf16* s_stage = (__bf16*)(smem_raw + 81920);   // 32768 B (8 waves x [16][128])
    float*  s_wbuf  = (float*)(smem_raw + 114688);   // 16384 B [16][256] logits
    float*  s_mi    = (float*)(smem_raw + 131072);   //  8192 B [16][128] m_i
    __bf16* s_cat   = (__bf16*)(smem_raw + 139264);  //  8192 B [16][256]
    __bf16* s_z1    = (__bf16*)(smem_raw + 147456);  //  4096 B [16][128]
    float*  s_hout  = (float*)(smem_raw + 151552);   //  8192 B [16][128]
    float*  s_b2    = (float*)(smem_raw + 159744);   //   512 B
    float*  s_bc1   = (float*)(smem_raw + 160256);   //   512 B
    float*  s_wc2   = (float*)(smem_raw + 160768);   //   512 B
    float*  s_mask  = (float*)(smem_raw + 161280);   //  1024 B
    // total 162304 B

    int b   = blockIdx.x >> 4;
    int i0  = (blockIdx.x & 15) << 4;
    int tid = threadIdx.x, lane = tid & 31, wave = tid >> 5;
    int col = lane & 15, hf = lane >> 4;

    // ---- stage weights / constants into LDS ----
    for (int x = tid; x < 512;  x += 256) ((uint4*)s_w1c)[x] = ((const uint4*)pkW1c)[x];
    for (int x = tid; x < 2048; x += 256) ((uint4*)s_wm2)[x] = ((const uint4*)pkWm2)[x];
    for (int x = tid; x < 2048; x += 256) ((uint4*)s_wc1)[x] = ((const uint4*)pkWc1)[x];
    for (int x = tid; x < 16 * HH; x += 256) {
        int m = x >> 7, n = x & 127;
        s_P[x]  = P[((b * NNODES) + i0 + m) * HH + n] + b_m1[n];
        s_mi[x] = 0.f;
    }
    for (int x = tid; x < HH; x += 256) {
        s_b2[x] = b_m2[x]; s_bc1[x] = b_c1[x]; s_wc2[x] = Wc2[x];
    }
    for (int x = tid; x < NNODES; x += 256) s_mask[x] = maskg[b * NNODES + x];
    __syncthreads();

    // per-lane constants
    float L[9];
    for (int c = 0; c < 9; c++) L[c] = latt[b * 9 + c];
    int   irow = i0 + col;                 // this lane's edge row (both halves)
    float fi0 = frac[((b * NNODES) + irow) * 3 + 0];
    float fi1 = frac[((b * NNODES) + irow) * 3 + 1];
    float fi2 = frac[((b * NNODES) + irow) * 3 + 2];
    float bc2v = b_c2[0];
    __bf16* st = s_stage + wave * (16 * HH);

    const float rbf_step  = 8.0f / 31.0f;
    const float rbf_gamma = 15.015625f;    // (31/8)^2

    v8f miacc[8];
    for (int nt = 0; nt < 8; nt++)
        for (int r = 0; r < 8; r++) miacc[nt][r] = 0.f;

    // ---- main edge loop: 32 j per wave ----
    #pragma unroll 1
    for (int jj = 0; jj < 32; jj++) {
        int j = wave * 32 + jj;

        // geometry (each half computes its own copy for row m)
        float d0 = fi0 - frac[((b * NNODES) + j) * 3 + 0]; d0 -= rintf(d0);
        float d1 = fi1 - frac[((b * NNODES) + j) * 3 + 1]; d1 -= rintf(d1);
        float d2 = fi2 - frac[((b * NNODES) + j) * 3 + 2]; d2 -= rintf(d2);
        float c0 = d0 * L[0] + d1 * L[3] + d2 * L[6];
        float c1 = d0 * L[1] + d1 * L[4] + d2 * L[7];
        float c2 = d0 * L[2] + d1 * L[5] + d2 * L[8];
        float dist = sqrtf(c0 * c0 + c1 * c1 + c2 * c2 + 1e-6f);

        // RBF directly in A-fragment layout (row m = lane&15)
        union { uint4 q[2]; v16bf v; } af;
        __bf16* ab = (__bf16*)&af;
        for (int e = 0; e < 16; e++) {
            int kb = (e < 8) ? (8 * hf + e) : (16 + 8 * hf + (e - 8));
            float t = dist - rbf_step * (float)kb;
            ab[e] = (__bf16)__expf(-rbf_gamma * t * t);
        }

        // GEMM1: x1 = rbf @ W1c, C preloaded with P_i + Q_j (+b1 in P)
        v8f x1[8];
        for (int nt = 0; nt < 8; nt++) {
            float qv = Q[((b * NNODES) + j) * HH + nt * 16 + col];
            for (int r = 0; r < 8; r++)
                x1[nt][r] = s_P[(r + 8 * hf) * HH + nt * 16 + col] + qv;
            x1[nt] = wmma_bf16(af.v, load_b_frag(s_w1c, nt, lane), x1[nt]);
        }
        // silu + stage for next GEMM (C-layout -> row-major bf16 tile)
        for (int nt = 0; nt < 8; nt++)
            for (int r = 0; r < 8; r++)
                st[(r + 8 * hf) * HH + nt * 16 + col] = (__bf16)silu_f(x1[nt][r]);

        // GEMM2: m = silu(x1) @ W_m2 + b2, then mask_j
        v8f mm[8];
        for (int nt = 0; nt < 8; nt++)
            for (int r = 0; r < 8; r++) mm[nt][r] = s_b2[nt * 16 + col];
        for (int kt = 0; kt < 4; kt++) {
            v16bf a = load_a_frag(st, HH, kt, lane);
            for (int nt = 0; nt < 8; nt++)
                mm[nt] = wmma_bf16(a, load_b_frag(s_wm2, nt * 4 + kt, lane), mm[nt]);
        }
        float mj = s_mask[j];
        for (int nt = 0; nt < 8; nt++)
            for (int r = 0; r < 8; r++) {
                float v = mm[nt][r] * mj;
                miacc[nt][r] += v;
                st[(r + 8 * hf) * HH + nt * 16 + col] = (__bf16)v;
            }

        // GEMM3: y = m @ W_c1 + bc1
        v8f y[8];
        for (int nt = 0; nt < 8; nt++)
            for (int r = 0; r < 8; r++) y[nt][r] = s_bc1[nt * 16 + col];
        for (int kt = 0; kt < 4; kt++) {
            v16bf a = load_a_frag(st, HH, kt, lane);
            for (int nt = 0; nt < 8; nt++)
                y[nt] = wmma_bf16(a, load_b_frag(s_wc1, nt * 4 + kt, lane), y[nt]);
        }

        // logit: w = silu(y) . Wc2 + bc2  (reduce over n via half-wave shfl)
        float p[8];
        for (int r = 0; r < 8; r++) p[r] = 0.f;
        for (int nt = 0; nt < 8; nt++) {
            float wv = s_wc2[nt * 16 + col];
            for (int r = 0; r < 8; r++) p[r] += silu_f(y[nt][r]) * wv;
        }
        for (int off = 1; off < 16; off <<= 1)
            for (int r = 0; r < 8; r++) p[r] += __shfl_xor(p[r], off, 32);
        if (col == 0)
            for (int r = 0; r < 8; r++)
                s_wbuf[(r + 8 * hf) * NNODES + j] = p[r] + bc2v;
    }

    // m_i cross-wave reduction
    for (int nt = 0; nt < 8; nt++)
        for (int r = 0; r < 8; r++)
            atomicAdd(&s_mi[(r + 8 * hf) * HH + nt * 16 + col], miacc[nt][r]);
    __syncthreads();

    // ---- softmax over j + coord update (16 threads per i-row) ----
    {
        int row = tid >> 4, sub = tid & 15;
        float mask_i = s_mask[i0 + row];
        int ig = (b * NNODES) + i0 + row;
        float g0 = frac[ig * 3 + 0], g1 = frac[ig * 3 + 1], g2 = frac[ig * 3 + 2];

        float mx = -1e30f;
        for (int t = 0; t < 16; t++) {
            int j = sub + 16 * t;
            float w = s_wbuf[row * NNODES + j];
            if (s_mask[j] * mask_i == 0.f) w = -1e9f;
            mx = fmaxf(mx, w);
        }
        for (int off = 1; off < 16; off <<= 1) mx = fmaxf(mx, __shfl_xor(mx, off, 32));

        float sum = 0.f, u0 = 0.f, u1 = 0.f, u2 = 0.f;
        for (int t = 0; t < 16; t++) {
            int j = sub + 16 * t;
            float w = s_wbuf[row * NNODES + j];
            if (s_mask[j] * mask_i == 0.f) w = -1e9f;
            float e = __expf(w - mx);
            sum += e;
            float d0 = g0 - frac[((b * NNODES) + j) * 3 + 0]; d0 -= rintf(d0);
            float d1 = g1 - frac[((b * NNODES) + j) * 3 + 1]; d1 -= rintf(d1);
            float d2 = g2 - frac[((b * NNODES) + j) * 3 + 2]; d2 -= rintf(d2);
            float c0 = d0 * L[0] + d1 * L[3] + d2 * L[6];
            float c1 = d0 * L[1] + d1 * L[4] + d2 * L[7];
            float c2 = d0 * L[2] + d1 * L[5] + d2 * L[8];
            float dist = sqrtf(c0 * c0 + c1 * c1 + c2 * c2 + 1e-6f);
            float inv = 1.0f / (dist + 1e-6f);
            u0 += e * c0 * inv; u1 += e * c1 * inv; u2 += e * c2 * inv;
        }
        for (int off = 1; off < 16; off <<= 1) {
            sum += __shfl_xor(sum, off, 32);
            u0  += __shfl_xor(u0,  off, 32);
            u1  += __shfl_xor(u1,  off, 32);
            u2  += __shfl_xor(u2,  off, 32);
        }
        if (sub == 0) {
            float inv = 1.0f / sum;
            out_coord[ig * 3 + 0] = fminf(fmaxf(u0 * inv, -0.5f), 0.5f) * mask_i;
            out_coord[ig * 3 + 1] = fminf(fmaxf(u1 * inv, -0.5f), 0.5f) * mask_i;
            out_coord[ig * 3 + 2] = fminf(fmaxf(u2 * inv, -0.5f), 0.5f) * mask_i;
        }
    }
    __syncthreads();

    // ---- node MLP: concat(h_i, m_i) @ W_n1 -> silu -> @ W_n2, +h, LayerNorm ----
    for (int x = tid; x < 16 * 256; x += 256) {
        int m = x >> 8, c = x & 255;
        float v = (c < HH) ? h[((b * NNODES) + i0 + m) * HH + c]
                           : s_mi[m * HH + (c - HH)];
        s_cat[x] = (__bf16)v;
    }
    __syncthreads();
    {
        int nt = wave;
        v8f acc;
        for (int r = 0; r < 8; r++) acc[r] = b_n1[nt * 16 + col];
        for (int kt = 0; kt < 8; kt++) {
            v16bf a = load_a_frag(s_cat, 256, kt, lane);
            acc = wmma_bf16(a, load_b_frag(pkWn1, nt * 8 + kt, lane), acc);
        }
        for (int r = 0; r < 8; r++)
            s_z1[(r + 8 * hf) * HH + nt * 16 + col] = (__bf16)silu_f(acc[r]);
    }
    __syncthreads();
    {
        int nt = wave;
        v8f acc;
        for (int r = 0; r < 8; r++) acc[r] = b_n2[nt * 16 + col];
        for (int kt = 0; kt < 4; kt++) {
            v16bf a = load_a_frag(s_z1, HH, kt, lane);
            acc = wmma_bf16(a, load_b_frag(pkWn2, nt * 4 + kt, lane), acc);
        }
        for (int r = 0; r < 8; r++) {
            int m = r + 8 * hf, n = nt * 16 + col;
            s_hout[m * HH + n] = h[((b * NNODES) + i0 + m) * HH + n] + acc[r];
        }
    }
    __syncthreads();
    {
        int row = tid >> 4, sub = tid & 15;
        float s1 = 0.f, s2 = 0.f;
        for (int k = 0; k < 8; k++) {
            int n = sub + 16 * k;
            float v = s_hout[row * HH + n];
            s1 += v; s2 += v * v;
        }
        for (int off = 1; off < 16; off <<= 1) {
            s1 += __shfl_xor(s1, off, 32);
            s2 += __shfl_xor(s2, off, 32);
        }
        float mu   = s1 * (1.0f / 128.0f);
        float var  = s2 * (1.0f / 128.0f) - mu * mu;
        float rinv = rsqrtf(var + 1e-5f);
        int ig = (b * NNODES) + i0 + row;
        for (int k = 0; k < 8; k++) {
            int n = sub + 16 * k;
            float v = s_hout[row * HH + n];
            out_h[ig * HH + n] = (v - mu) * rinv * ln_g[n] + ln_b[n];
        }
    }
}

// ---------------------------------------------------------------------------
// Host launch
// ---------------------------------------------------------------------------
extern "C" void kernel_launch(void* const* d_in, const int* in_sizes, int n_in,
                              void* d_out, int out_size, void* d_ws, size_t ws_size,
                              hipStream_t stream) {
    const float* h    = (const float*)d_in[0];
    const float* frac = (const float*)d_in[1];
    const float* latt = (const float*)d_in[2];
    const float* mask = (const float*)d_in[3];
    const float* W_m1 = (const float*)d_in[4];
    const float* b_m1 = (const float*)d_in[5];
    const float* W_m2 = (const float*)d_in[6];
    const float* b_m2 = (const float*)d_in[7];
    const float* W_c1 = (const float*)d_in[8];
    const float* b_c1 = (const float*)d_in[9];
    const float* W_c2 = (const float*)d_in[10];
    const float* b_c2 = (const float*)d_in[11];
    const float* W_n1 = (const float*)d_in[12];
    const float* b_n1 = (const float*)d_in[13];
    const float* W_n2 = (const float*)d_in[14];
    const float* b_n2 = (const float*)d_in[15];
    const float* ln_g = (const float*)d_in[16];
    const float* ln_b = (const float*)d_in[17];
    (void)in_sizes; (void)n_in; (void)out_size; (void)ws_size;

    // workspace layout (bf16 elements for packed weights, 256 KB reserved)
    __bf16* pk = (__bf16*)d_ws;
    const int OFF_W1A = 0;        // 32 frags
    const int OFF_W1B = 16384;    // 32 frags
    const int OFF_W1C = 32768;    //  8 frags
    const int OFF_WM2 = 36864;    // 32 frags
    const int OFF_WC1 = 53248;    // 32 frags
    const int OFF_WN1 = 69632;    // 64 frags
    const int OFF_WN2 = 102400;   // 32 frags  (ends at 118784 elems)
    float* Pws = (float*)((char*)d_ws + 256 * 1024);
    float* Qws = Pws + BB * NNODES * HH;

    // 1) pack weights into bf16 fragment layout
    pack_w_kernel<<<64, 256, 0, stream>>>(W_m1,             pk + OFF_W1A, 128, 128);
    pack_w_kernel<<<64, 256, 0, stream>>>(W_m1 + 128 * 128, pk + OFF_W1B, 128, 128);
    pack_w_kernel<<<16, 256, 0, stream>>>(W_m1 + 256 * 128, pk + OFF_W1C,  32, 128);
    pack_w_kernel<<<64, 256, 0, stream>>>(W_m2,             pk + OFF_WM2, 128, 128);
    pack_w_kernel<<<64, 256, 0, stream>>>(W_c1,             pk + OFF_WC1, 128, 128);
    pack_w_kernel<<<128, 256, 0, stream>>>(W_n1,            pk + OFF_WN1, 256, 128);
    pack_w_kernel<<<64, 256, 0, stream>>>(W_n2,             pk + OFF_WN2, 128, 128);

    // 2) P = h @ W1a, Q = h @ W1b
    pq_kernel<<<BB * (NNODES / 16), 256, 16 * HH * 2, stream>>>(
        h, pk + OFF_W1A, pk + OFF_W1B, Pws, Qws);

    // 3) fused edge + attention + node MLP + LN
    const int SMEM = 162304;
    hipFuncSetAttribute((const void*)egnn_edge_kernel,
                        hipFuncAttributeMaxDynamicSharedMemorySize, SMEM);
    float* out_h     = (float*)d_out;
    float* out_coord = out_h + BB * NNODES * HH;
    egnn_edge_kernel<<<BB * (NNODES / 16), 256, SMEM, stream>>>(
        h, frac, latt, mask, b_m1, b_m2, b_c1, W_c2, b_c2, b_n1, b_n2,
        ln_g, ln_b,
        pk + OFF_W1C, pk + OFF_WM2, pk + OFF_WC1, pk + OFF_WN1, pk + OFF_WN2,
        Pws, Qws, out_h, out_coord);
}
